// SamplingBlock_38362647887990
// MI455X (gfx1250) — compile-verified, matrix operand order checked by reading, not run on hardware
//
#include <hip/hip_runtime.h>
#include <hip/hip_bf16.h>

typedef __attribute__((ext_vector_type(16))) _Float16 v16h;
typedef __attribute__((ext_vector_type(8)))  _Float16 v8h;
typedef __attribute__((ext_vector_type(8)))  float    v8f;

#define CFEAT 256
#define NNEIGH 8
#define NVERT 8192
#define NBATCH 4
#define GRID 32
#define VOL (GRID*GRID*GRID)
#define KSEG 2336          // 9*256 feature slots + 27 coords + 5 pad (multiple of 32)
#define NCHUNK 73          // KSEG / 32
#define MPTS 32            // points per block (M tile)
#define WPACK_HALVES ((size_t)16 * NCHUNK * 32 * 16)   // o_tiles * chunks * lanes * halves
#define XT_HALVES ((size_t)NBATCH * VOL * CFEAT)       // transposed f16 volume

// ---------------------------------------------------------------------------
// Trilinear corner setup (matches grid_sample bilinear/border/align_corners)
// ---------------------------------------------------------------------------
struct Corners { int o[8]; float w[8]; };

__device__ inline Corners corner_setup(float cx, float cy, float cz) {
    Corners r;
    float gx = fminf(fmaxf((cx + 1.f) * 0.5f * (GRID - 1), 0.f), (float)(GRID - 1));
    float gy = fminf(fmaxf((cy + 1.f) * 0.5f * (GRID - 1), 0.f), (float)(GRID - 1));
    float gz = fminf(fmaxf((cz + 1.f) * 0.5f * (GRID - 1), 0.f), (float)(GRID - 1));
    float x0f = floorf(gx), y0f = floorf(gy), z0f = floorf(gz);
    float fx = gx - x0f, fy = gy - y0f, fz = gz - z0f;
    int x0 = (int)x0f, y0 = (int)y0f, z0 = (int)z0f;
    int x1 = min(x0 + 1, GRID - 1);
    int y1 = min(y0 + 1, GRID - 1);
    int z1 = min(z0 + 1, GRID - 1);
    float gxw = 1.f - fx, gyw = 1.f - fy, gzw = 1.f - fz;
    r.o[0] = (z0 * GRID + y0) * GRID + x0;  r.w[0] = gxw * gyw * gzw;
    r.o[1] = (z0 * GRID + y0) * GRID + x1;  r.w[1] = fx  * gyw * gzw;
    r.o[2] = (z0 * GRID + y1) * GRID + x0;  r.w[2] = gxw * fy  * gzw;
    r.o[3] = (z0 * GRID + y1) * GRID + x1;  r.w[3] = fx  * fy  * gzw;
    r.o[4] = (z1 * GRID + y0) * GRID + x0;  r.w[4] = gxw * gyw * fz;
    r.o[5] = (z1 * GRID + y0) * GRID + x1;  r.w[5] = fx  * gyw * fz;
    r.o[6] = (z1 * GRID + y1) * GRID + x0;  r.w[6] = gxw * fy  * fz;
    r.o[7] = (z1 * GRID + y1) * GRID + x1;  r.w[7] = fx  * fy  * fz;
    return r;
}

// Accumulate 32 channels of a trilinear sample from the channel-contiguous
// f16 volume: 8 corners x 4 coalesced b128 loads each.
__device__ inline void sample_accum(const _Float16* __restrict__ xtb,
                                    const Corners& cr, int c0, float* acc) {
    #pragma unroll
    for (int i = 0; i < 32; ++i) acc[i] = 0.f;
    #pragma unroll
    for (int k = 0; k < 8; ++k) {
        const v8h* s = (const v8h*)(xtb + (size_t)cr.o[k] * CFEAT + c0);
        const float w = cr.w[k];
        #pragma unroll
        for (int q = 0; q < 4; ++q) {
            v8h v = s[q];
            #pragma unroll
            for (int i = 0; i < 8; ++i) acc[q * 8 + i] += w * (float)v[i];
        }
    }
}

__device__ inline void store32_f16(_Float16* dst, const float* acc) {
    #pragma unroll
    for (int q = 0; q < 4; ++q) {
        v8h o;
        #pragma unroll
        for (int i = 0; i < 8; ++i) o[i] = (_Float16)acc[q * 8 + i];
        *(v8h*)(dst + q * 8) = o;
    }
}

__device__ inline v16h load_afrag(const _Float16* row, int kbase, int hi) {
    v8h lo = *(const v8h*)(row + kbase + hi);        // K 0..7   (or 8..15)
    v8h hh = *(const v8h*)(row + kbase + 16 + hi);   // K 16..23 (or 24..31)
    v16h a;
    #pragma unroll
    for (int i = 0; i < 8; ++i) { a[i] = lo[i]; a[i + 8] = hh[i]; }
    return a;
}

// ---------------------------------------------------------------------------
// Kernel 0: tiled transpose  x[B,C,32^3] f32  ->  xT[B,32^3,C] f16.
// 64-voxel x 256-channel tile via LDS; coalesced reads and writes.
// ---------------------------------------------------------------------------
#define TP_V 64
__global__ __launch_bounds__(256) void transpose_vol(
    const float* __restrict__ x, _Float16* __restrict__ xT)
{
    __shared__ _Float16 tile[TP_V][CFEAT + 8];   // +8 halves pad vs bank conflicts
    const int nb = VOL / TP_V;                   // 512
    const int b  = blockIdx.x / nb;
    const int vb = (blockIdx.x % nb) * TP_V;
    const int vi = threadIdx.x & 63;
    const int cc = threadIdx.x >> 6;             // 0..3
    const float* __restrict__ xb = x + (size_t)b * CFEAT * VOL;
    for (int r = 0; r < 64; ++r) {
        int c = cc * 64 + r;
        tile[vi][c] = (_Float16)xb[(size_t)c * VOL + vb + vi];   // coalesced over vi
    }
    __syncthreads();
    const int v = threadIdx.x >> 2;
    const int q = threadIdx.x & 3;
    _Float16* dst = xT + ((size_t)b * VOL + vb + v) * CFEAT + q * 64;
    #pragma unroll
    for (int t8 = 0; t8 < 8; ++t8) {
        v8h val;
        #pragma unroll
        for (int i = 0; i < 8; ++i) val[i] = tile[v][q * 64 + t8 * 8 + i];
        *(v8h*)(dst + t8 * 8) = val;             // 128B contiguous per thread
    }
}

// ---------------------------------------------------------------------------
// Kernel 1: fold W_sum @ W_diff (+ W_center into the k=0 slice) directly into
// the WMMA B-fragment lane layout (mirror of documented 16-bit A 16x32 layout).
// ---------------------------------------------------------------------------
__global__ __launch_bounds__(512) void pack_weights(
    const float* __restrict__ Wdiff,    // [256, 259]
    const float* __restrict__ Wcenter,  // [256, 259]
    const float* __restrict__ Wsum,     // [256, 256, 9]
    _Float16* __restrict__ Wpack)
{
    const int chunk = blockIdx.x;        // 0..72
    const int otile = blockIdx.y;        // 0..15
    const int lane  = threadIdx.x & 31;
    const int h     = threadIdx.x >> 5;  // 0..15
    const int o     = otile * 16 + (lane & 15);
    const int klocal = ((h >> 3) << 4) | (h & 7) | ((lane & 16) ? 8 : 0);
    const int kk = chunk * 32 + klocal;

    float v = 0.f;
    int kseg = 0, cp = 0;
    bool valid = true;
    if (kk < 9 * 256)         { kseg = kk >> 8; cp = kk & 255; }
    else if (kk < 9*256 + 27) { int t = kk - 9 * 256; kseg = t / 3; cp = 256 + t % 3; }
    else                      { valid = false; }

    if (valid) {
        float s = 0.f;
        for (int c = 0; c < 256; ++c)
            s += Wsum[(o * 256 + c) * 9 + kseg] * Wdiff[c * 259 + cp];
        if (kseg == 0) s += Wcenter[o * 259 + cp];
        v = s;
    }
    Wpack[(((size_t)otile * NCHUNK + chunk) * 32 + lane) * 16 + h] = (_Float16)v;
}

// ---------------------------------------------------------------------------
// Kernel 2: fold biases: btot[o] = b_sum + b_center + sum_{k,c} W_sum*b_diff
// ---------------------------------------------------------------------------
__global__ __launch_bounds__(256) void bias_fold(
    const float* __restrict__ bdiff, const float* __restrict__ bcenter,
    const float* __restrict__ Wsum,  const float* __restrict__ bsum,
    float* __restrict__ btot)
{
    int o = threadIdx.x;
    float s = bsum[o] + bcenter[o];
    for (int c = 0; c < 256; ++c) {
        float bd = bdiff[c];
        for (int k = 0; k < 9; ++k)
            s += Wsum[(o * 256 + c) * 9 + k] * bd;
    }
    btot[o] = s;
}

// ---------------------------------------------------------------------------
// Kernel 3: fused sample + shift + neighbour sample + WMMA GEMM.
// One block = 32 points (M tile), 256 threads = 8 waves.
// Wave w: output channels [32w, 32w+32) x 32 points -> 4 WMMA accumulators.
// ---------------------------------------------------------------------------
__global__ __launch_bounds__(256) void fused_sampling_block(
    const _Float16* __restrict__ xT,   // [B, 32^3, C] f16 (channel-contiguous)
    const float* __restrict__ verts,   // [B, N, 3]
    const float* __restrict__ Wshift,  // [24, 256]
    const float* __restrict__ bshift,  // [24]
    const _Float16* __restrict__ Wpack,
    const float* __restrict__ btot,
    float* __restrict__ out)           // [B*N, 256]
{
    __shared__ __align__(32) _Float16 feat[MPTS][KSEG];   // ~149.5 KB
    __shared__ float nco[MPTS][NNEIGH * 3];               // neighbour coords

    const int tid   = threadIdx.x;
    const int gbase = blockIdx.x * MPTS;
    const int p     = tid >> 3;          // point 0..31
    const int c0    = (tid & 7) << 5;    // 32-channel group
    const int g     = gbase + p;
    const int b     = g >> 13;           // / 8192
    const float vx = verts[g * 3 + 0];
    const float vy = verts[g * 3 + 1];
    const float vz = verts[g * 3 + 2];
    const _Float16* __restrict__ xtb = xT + (size_t)b * VOL * CFEAT;

    float acc[32];

    // ---- 1. sample xp at vertex (coalesced f16 vector gathers)
    {
        Corners cr = corner_setup(vx, vy, vz);
        sample_accum(xtb, cr, c0, acc);
        store32_f16(&feat[p][c0], acc);
    }
    __syncthreads();

    // ---- 2. shift = W_shift @ xp + b_shift ; neigh = vert + shift
    for (int task = tid; task < MPTS * 24; task += 256) {
        int pp = task / 24, j = task % 24;       // j = neighbour*3 + dim
        float s = bshift[j];
        const float* wr = Wshift + j * 256;
        for (int c = 0; c < 256; ++c) s += wr[c] * (float)feat[pp][c];
        float coord = verts[(gbase + pp) * 3 + (j % 3)] + s;
        nco[pp][j] = coord;
        feat[pp][9 * 256 + (j / 3 + 1) * 3 + (j % 3)] = (_Float16)coord;
    }
    for (int task = tid; task < MPTS * 8; task += 256) {
        int pp = task >> 3, d = task & 7;
        if (d < 3) feat[pp][9 * 256 + d] = (_Float16)verts[(gbase + pp) * 3 + d];
        else       feat[pp][9 * 256 + 27 + (d - 3)] = (_Float16)0.f;
    }
    __syncthreads();

    // ---- 3. sample the 8 neighbours
    for (int j = 0; j < NNEIGH; ++j) {
        Corners cr = corner_setup(nco[p][j * 3 + 0], nco[p][j * 3 + 1], nco[p][j * 3 + 2]);
        sample_accum(xtb, cr, c0, acc);
        store32_f16(&feat[p][(j + 1) * 256 + c0], acc);
    }
    __syncthreads();

    // ---- 4. WMMA GEMM: 32 points x 256 outputs, K = 2336.
    const int lane = tid & 31;
    const int wv   = tid >> 5;
    const int m    = lane & 15;
    const int hi   = (lane & 16) ? 8 : 0;
    v8f a00 = {}, a01 = {}, a10 = {}, a11 = {};
    const _Float16* __restrict__ wp0 = Wpack + (size_t)(2 * wv)     * NCHUNK * 512;
    const _Float16* __restrict__ wp1 = Wpack + (size_t)(2 * wv + 1) * NCHUNK * 512;

    for (int ch = 0; ch < NCHUNK; ++ch) {
        const int kbase = ch * 32;
        v16h af0 = load_afrag(&feat[m][0],      kbase, hi);   // points 0..15
        v16h af1 = load_afrag(&feat[m + 16][0], kbase, hi);   // points 16..31
        v16h bf0 = *(const v16h*)(wp0 + ((size_t)(kbase + lane)) * 16);
        v16h bf1 = *(const v16h*)(wp1 + ((size_t)(kbase + lane)) * 16);
        a00 = __builtin_amdgcn_wmma_f32_16x16x32_f16(false, af0, false, bf0,
                                                     (short)0, a00, false, false);
        a01 = __builtin_amdgcn_wmma_f32_16x16x32_f16(false, af0, false, bf1,
                                                     (short)0, a01, false, false);
        a10 = __builtin_amdgcn_wmma_f32_16x16x32_f16(false, af1, false, bf0,
                                                     (short)0, a10, false, false);
        a11 = __builtin_amdgcn_wmma_f32_16x16x32_f16(false, af1, false, bf1,
                                                     (short)0, a11, false, false);
    }

    // ---- 5. epilogue: + btot, scatter per documented C/D layout
    const int nloc = lane & 15;
    const int mofs = (lane & 16) ? 8 : 0;
    const float bt0 = btot[wv * 32 + nloc];
    const float bt1 = btot[wv * 32 + 16 + nloc];
    #pragma unroll
    for (int r = 0; r < 8; ++r) {
        size_t row0 = (size_t)(gbase + r + mofs) * 256;
        size_t row1 = (size_t)(gbase + 16 + r + mofs) * 256;
        out[row0 + wv * 32 + nloc]      = a00[r] + bt0;
        out[row0 + wv * 32 + 16 + nloc] = a01[r] + bt1;
        out[row1 + wv * 32 + nloc]      = a10[r] + bt0;
        out[row1 + wv * 32 + 16 + nloc] = a11[r] + bt1;
    }
}

// ---------------------------------------------------------------------------
extern "C" void kernel_launch(void* const* d_in, const int* in_sizes, int n_in,
                              void* d_out, int out_size, void* d_ws, size_t ws_size,
                              hipStream_t stream) {
    const float* x       = (const float*)d_in[0];
    const float* verts   = (const float*)d_in[1];
    const float* Wshift  = (const float*)d_in[2];
    const float* bshift  = (const float*)d_in[3];
    const float* Wdiff   = (const float*)d_in[4];
    const float* bdiff   = (const float*)d_in[5];
    const float* Wcenter = (const float*)d_in[6];
    const float* bcenter = (const float*)d_in[7];
    const float* Wsum    = (const float*)d_in[8];
    const float* bsum    = (const float*)d_in[9];
    float* out = (float*)d_out;

    // workspace layout: xT (64 MB) | Wpack (1.17 MB) | btot (1 KB)
    _Float16* xT    = (_Float16*)d_ws;
    _Float16* Wpack = (_Float16*)((char*)d_ws + XT_HALVES * sizeof(_Float16));
    float*    btot  = (float*)((char*)Wpack + WPACK_HALVES * sizeof(_Float16));

    transpose_vol<<<NBATCH * (VOL / TP_V), 256, 0, stream>>>(x, xT);
    pack_weights<<<dim3(NCHUNK, 16), 512, 0, stream>>>(Wdiff, Wcenter, Wsum, Wpack);
    bias_fold<<<1, 256, 0, stream>>>(bdiff, bcenter, Wsum, bsum, btot);
    fused_sampling_block<<<(NBATCH * NVERT) / MPTS, 256, 0, stream>>>(
        xT, verts, Wshift, bshift, Wpack, btot, out);
}
